// VballNetV1d_80951543595269
// MI455X (gfx1250) — matrix operand order repaired and checked by reading, not compiled, not run on hardware
//
#include <hip/hip_runtime.h>
#include <hip/hip_bf16.h>
#include <math.h>

typedef __attribute__((ext_vector_type(16))) _Float16 v16h;
typedef __attribute__((ext_vector_type(8)))  float    v8f;

// ---------------------------------------------------------------------------
// Implicit-GEMM 2D conv (SAME, KS=3 or 1) via V_WMMA_F32_16X16X32_F16.
// M = output pixels (n,h,w flattened), N = output channels, K = Cin*KS*KS.
// Each wave computes 4 M-subtiles x 16 couts, reusing the weight fragment.
// All gathers are UNCONDITIONAL loads from clamped addresses; validity is
// applied afterwards with selects (no exec-divergent load-per-element).
// All offset arithmetic is 32-bit (every tensor < 2^31 elements).
// Implicit concat/upsample on input channels:
//   c <  C0 : in0 sampled at (h>>s0, w>>s0)   (s0=1 -> nearest-upsample x2)
//   c >= C0 : in1 at full resolution
// ---------------------------------------------------------------------------
template <int KS>
__global__ __launch_bounds__(256)
void conv_wmma_kernel(const float* __restrict__ in0, int C0, int s0,
                      const float* __restrict__ in1, int C1,
                      const float* __restrict__ wt,
                      const float* __restrict__ bias, int act,
                      float* __restrict__ out,
                      int N, int H, int wshift, int Cout)
{
    constexpr int KSQ = KS * KS;
    constexpr int PAD = KS / 2;
    constexpr int MT  = 4;                 // M-subtiles per wave
    const int W    = 1 << wshift;
    const int HW   = H << wshift;
    const int Mtot = N * HW;
    const int lane = threadIdx.x & 31;
    const int wid  = threadIdx.x >> 5;
    const int base = (blockIdx.x * 8 + wid) * (16 * MT);
    if (base >= Mtot) return;              // wave-uniform exit (EXEC stays full)

    const int Cin  = C0 + C1;
    const int Ktot = Cin * KSQ;
    const int Hs   = H >> s0;              // in0 height
    const int wsh0 = wshift - s0;          // in0 width shift
    const int m    = lane & 15;            // A-row within subtile
    const int co   = blockIdx.y * 16 + (lane & 15);
    const bool cvalid = (co < Cout);
    const int coc = cvalid ? co : (Cout - 1);
    const int kbaseA = (lane < 16) ? 0 : 16;  // A halves j -> K = base + j
    const int kbaseB = (lane < 16) ? 0 : 16;  // B halves j -> K = base + j
    const float* __restrict__ wrow = wt + coc * Ktot;

    // Each 16-px subtile lies in one image row (W % 16 == 0, 16-aligned tiles).
    // Decode once per subtile; clamp dead subtiles to a valid pixel so all
    // later address math stays in-bounds (their lanes are never stored).
    int nn[MT], hh[MT], wb[MT];
    bool tv[MT];
#pragma unroll
    for (int t = 0; t < MT; ++t) {
        int pt = base + t * 16;
        tv[t] = (pt < Mtot);
        int ptc = tv[t] ? pt : (Mtot - 16);
        int row = ptc >> wshift;           // n*H + h
        wb[t] = ptc & (W - 1);
        hh[t] = row % H;
        nn[t] = row / H;
    }

    v8f acc[MT] = {};
    for (int kk = 0; kk < Ktot; kk += 32) {
        if (kk + 32 < Ktot)
            __builtin_prefetch(wrow + kk + 32, 0, 0);   // global_prefetch_b8

        // ---- B fragment (weights): unconditional clamped loads, then mask ----
        float vb[16];
#pragma unroll
        for (int j = 0; j < 16; ++j) {
            int kb = kk + kbaseB + j;
            vb[j] = wrow[kb < Ktot ? kb : (Ktot - 1)];
        }
        v16h b;
#pragma unroll
        for (int j = 0; j < 16; ++j) {
            int kb = kk + kbaseB + j;
            float f = (cvalid && kb < Ktot) ? vb[j] : 0.f;
            b[j] = (_Float16)f;
        }

#pragma unroll
        for (int t = 0; t < MT; ++t) {
            float va[16];
            bool  ok[16];
#pragma unroll
            for (int j = 0; j < 16; ++j) {
                int ka = kk + kbaseA + j;
                bool kin = (ka < Ktot);
                int kac = kin ? ka : (Ktot - 1);
                int c  = kac / KSQ;                 // constant-divisor
                int rr = kac - c * KSQ;
                int kh = rr / KS, kw = rr - kh * KS;
                int ih = hh[t] + kh - PAD;
                int iw = wb[t] + m + kw - PAD;
                ok[j] = kin && (ih >= 0) && (ih < H) && (iw >= 0) && (iw < W);
                int ihc = ih < 0 ? 0 : (ih >= H ? H - 1 : ih);
                int iwc = iw < 0 ? 0 : (iw >= W ? W - 1 : iw);
                bool use0 = (c < C0);
                // branchless 32-bit address + pointer select
                int off0 = (((nn[t] * C0 + (use0 ? c : 0)) * Hs + (ihc >> s0)) << wsh0)
                           + (iwc >> s0);
                int off1 = (((nn[t] * C1 + (use0 ? 0 : c - C0)) * H + ihc) << wshift)
                           + iwc;
                const float* p = use0 ? in0 : in1;
                int off = use0 ? off0 : off1;
                va[j] = p[off];            // unconditional load (address valid)
            }
            v16h a;
#pragma unroll
            for (int j = 0; j < 16; ++j)
                a[j] = (_Float16)(ok[j] ? va[j] : 0.f);
            acc[t] = __builtin_amdgcn_wmma_f32_16x16x32_f16(
                         false, a, false, b, (short)0, acc[t], false, false);
        }
    }

    const float bs = (bias != nullptr && cvalid) ? bias[coc] : 0.f;
#pragma unroll
    for (int t = 0; t < MT; ++t) {
        if (!tv[t] || !cvalid) continue;
        const int obase = (((nn[t] * Cout + co) * H + hh[t]) << wshift) + wb[t];
#pragma unroll
        for (int r = 0; r < 8; ++r) {
            int mm = r + ((lane >> 4) << 3);   // D layout: VGPR r -> M=r (+8 hi lanes)
            float v = acc[t][r] + bs;
            if (act == 1)      v = fmaxf(v, 0.f);
            else if (act == 2) v = 1.f / (1.f + __expf(-v));
            out[obase + mm] = v;
        }
    }
}

// ---------------------------------------------------------------------------
// GEMM  C(M,N) = A(M,K) * B(N,K)^T   (row-major f32 in/out, f16 WMMA compute)
// REQUIRES: K % 32 == 0, M % 16 == 0, N % 16 == 0 (no partial tiles -> no
// store guards, straight-line scale_offset stores). All loads unconditional.
// ---------------------------------------------------------------------------
__global__ __launch_bounds__(256)
void gemm_nt_wmma_kernel(const float* __restrict__ A, const float* __restrict__ B,
                         float* __restrict__ Cd, int M, int N, int K)
{
    const int lane = threadIdx.x & 31, wid = threadIdx.x >> 5;
    const int tm = blockIdx.x * 8 + wid, tn = blockIdx.y;
    if (tm * 16 >= M) return;
    const int row = tm * 16 + (lane & 15);
    const int col = tn * 16 + (lane & 15);
    const float* __restrict__ Ar = A + row * K;
    const float* __restrict__ Br = B + col * K;
    const int kbA = (lane < 16) ? 0 : 16;
    const int kbB = (lane < 16) ? 0 : 16;
    v8f acc = {};
    for (int kk = 0; kk < K; kk += 32) {
        float va[16], vb[16];
#pragma unroll
        for (int j = 0; j < 16; ++j) {
            va[j] = Ar[kk + kbA + j];
            vb[j] = Br[kk + kbB + j];
        }
        v16h a, b;
#pragma unroll
        for (int j = 0; j < 16; ++j) {
            a[j] = (_Float16)va[j];
            b[j] = (_Float16)vb[j];
        }
        acc = __builtin_amdgcn_wmma_f32_16x16x32_f16(
                  false, a, false, b, (short)0, acc, false, false);
    }
    const int mrow = tm * 16 + ((lane >> 4) << 3);
#pragma unroll
    for (int r = 0; r < 8; ++r)
        Cd[(mrow + r) * N + col] = acc[r];
}

// ---------------------------------------------------------------------------
// BatchNorm training-mode: per-channel biased stats over (N,H,W).
// ---------------------------------------------------------------------------
__global__ __launch_bounds__(256)
void bn_stats_kernel(const float* __restrict__ x, int N, int C, int HW,
                     float* __restrict__ mean, float* __restrict__ rstd)
{
    const int c = blockIdx.x;
    float s = 0.f, s2 = 0.f;
    for (int n = 0; n < N; ++n) {
        const float* bp = x + (n * C + c) * HW;
        for (int i = threadIdx.x; i < HW; i += 256) {
            float v = bp[i];
            s += v; s2 += v * v;
        }
    }
    __shared__ float ls[256], lq[256];
    ls[threadIdx.x] = s; lq[threadIdx.x] = s2;
    __syncthreads();
    for (int st = 128; st > 0; st >>= 1) {
        if ((int)threadIdx.x < st) {
            ls[threadIdx.x] += ls[threadIdx.x + st];
            lq[threadIdx.x] += lq[threadIdx.x + st];
        }
        __syncthreads();
    }
    if (threadIdx.x == 0) {
        const float per = (float)N * (float)HW;
        float mu  = ls[0] / per;
        float var = lq[0] / per - mu * mu;
        mean[c] = mu;
        rstd[c] = rsqrtf(var + 1e-5f);
    }
}

// grid = (ceil(HW/256), C, N): channel from blockIdx.y, no division.
__global__ void bn_apply_relu_kernel(float* __restrict__ x, int C, int HW,
                                     const float* __restrict__ mean,
                                     const float* __restrict__ rstd,
                                     const float* __restrict__ g,
                                     const float* __restrict__ b)
{
    int hw = blockIdx.x * 256 + threadIdx.x;
    if (hw >= HW) return;
    int c = blockIdx.y;
    int i = (blockIdx.z * C + c) * HW + hw;
    float v = (x[i] - mean[c]) * rstd[c] * g[c] + b[c];
    x[i] = fmaxf(v, 0.f);
}

// grid = (ceil(Ho*Wo/256), N*C); Wo power of two.
__global__ void maxpool2_kernel(const float* __restrict__ in, float* __restrict__ out,
                                int Ho, int woShift)
{
    const int Wo = 1 << woShift;
    int i = blockIdx.x * 256 + threadIdx.x;
    if (i >= (Ho << woShift)) return;
    int nc = blockIdx.y;
    int wo = i & (Wo - 1), ho = i >> woShift;
    const int Wi = Wo * 2;
    const float* bp = in + (nc * (Ho * 2) + ho * 2) * Wi + wo * 2;
    out[((nc * Ho) << woShift) + i] =
        fmaxf(fmaxf(bp[0], bp[1]), fmaxf(bp[Wi], bp[Wi + 1]));
}

// top-2 of heat per batch (tie-break: lowest index, matching lax.top_k)
__global__ __launch_bounds__(256)
void top2_kernel(const float* __restrict__ heat, int HW, int Wp, int* __restrict__ hwq)
{
    const int b = blockIdx.x;
    const float* hb = heat + b * HW;
    __shared__ float sv[256];
    __shared__ int   si[256];
    int best0 = -1;
    for (int pass = 0; pass < 2; ++pass) {
        float bv = -1e30f; int bi = 0x7fffffff;
        for (int i = threadIdx.x; i < HW; i += 256) {
            float v = hb[i];
            bool skip = (pass == 1 && i == best0);
            if (!skip && (v > bv || (v == bv && i < bi))) { bv = v; bi = i; }
        }
        sv[threadIdx.x] = bv; si[threadIdx.x] = bi;
        __syncthreads();
        for (int st = 128; st > 0; st >>= 1) {
            if ((int)threadIdx.x < st) {
                float ov = sv[threadIdx.x + st]; int oi = si[threadIdx.x + st];
                if (ov > sv[threadIdx.x] || (ov == sv[threadIdx.x] && oi < si[threadIdx.x])) {
                    sv[threadIdx.x] = ov; si[threadIdx.x] = oi;
                }
            }
            __syncthreads();
        }
        int win = si[0];
        __syncthreads();
        if (pass == 0) best0 = win;
        else if (threadIdx.x == 0) {
            hwq[b * 4 + 0] = best0 / Wp; hwq[b * 4 + 1] = best0 % Wp;
            hwq[b * 4 + 2] = win   / Wp; hwq[b * 4 + 3] = win   % Wp;
        }
    }
}

// nodes[(b*18 + t*2 + k), c] = xb[b, c, hh[b,k], ww[b,k]]  (identical across t)
__global__ void gather_nodes_kernel(const float* __restrict__ xb, const int* __restrict__ hwq,
                                    float* __restrict__ nodes, int C, int Hp, int Wp)
{
    int i = blockIdx.x * blockDim.x + threadIdx.x;
    if (i >= 144 * 256) return;
    int c = i & 255, node = i >> 8;
    int b = node / 18, k = node & 1;
    int hh = hwq[b * 4 + k * 2], ww = hwq[b * 4 + k * 2 + 1];
    nodes[i] = xb[((b * C + c) * Hp + hh) * Wp + ww];
}

// per-(node,head) attention coefficients: as = <h, a_s>, ad = <h, a_d>
__global__ void gat_coef_kernel(const float* __restrict__ h, const float* __restrict__ a_s,
                                const float* __restrict__ a_d,
                                float* __restrict__ asb, float* __restrict__ adb)
{
    int i = blockIdx.x * blockDim.x + threadIdx.x;  // 144*4
    if (i >= 144 * 4) return;
    int head = i & 3, n = i >> 2;
    float s = 0.f, d = 0.f;
    for (int j = 0; j < 64; ++j) {
        float hv = h[n * 256 + head * 64 + j];
        s += hv * a_s[head * 64 + j];
        d += hv * a_d[head * 64 + j];
    }
    asb[i] = s; adb[i] = d;
}

__device__ __forceinline__ float lrelu02(float x) { return x > 0.f ? x : 0.2f * x; }

// Edge pattern (replicated reference bug): node n receives self (x1),
// n-2 (x8) if 2<=n<=17, n+2 (x8) if n<=15. Segment softmax + aggregate + bias + relu.
__global__ void gat_aggregate_kernel(const float* __restrict__ h,
                                     const float* __restrict__ asb,
                                     const float* __restrict__ adb,
                                     const float* __restrict__ bias,
                                     float* __restrict__ out)
{
    int i = blockIdx.x * blockDim.x + threadIdx.x;  // 144*256
    if (i >= 144 * 256) return;
    int c = i & 255, n = i >> 8;
    int head = c >> 6;
    float ad = adb[n * 4 + head];
    float e0 = lrelu02(asb[n * 4 + head] + ad);
    bool hasPrev = (n >= 2 && n <= 17);
    bool hasNext = (n <= 15);
    float e1 = hasPrev ? lrelu02(asb[(n - 2) * 4 + head] + ad) : -1e30f;
    float e2 = hasNext ? lrelu02(asb[(n + 2) * 4 + head] + ad) : -1e30f;
    float mx = fmaxf(e0, fmaxf(e1, e2));
    float w0 = __expf(e0 - mx);
    float w1 = hasPrev ? 8.f * __expf(e1 - mx) : 0.f;
    float w2 = hasNext ? 8.f * __expf(e2 - mx) : 0.f;
    float acc = w0 * h[n * 256 + c];
    if (hasPrev) acc += w1 * h[(n - 2) * 256 + c];
    if (hasNext) acc += w2 * h[(n + 2) * 256 + c];
    out[i] = fmaxf(acc / (w0 + w1 + w2) + bias[c], 0.f);
}

// xb[b, c, hh, ww] += g[node = b*18 + 16 + k, c]   (t = 8 wins, per reference)
__global__ void scatter_add_kernel(const float* __restrict__ g, const int* __restrict__ hwq,
                                   float* __restrict__ xb, int C, int Hp, int Wp)
{
    int i = blockIdx.x * blockDim.x + threadIdx.x;  // 16*256
    if (i >= 16 * 256) return;
    int c = i & 255, bk = i >> 8;
    int b = bk >> 1, k = bk & 1;
    int node = b * 18 + 16 + k;
    int hh = hwq[b * 4 + k * 2], ww = hwq[b * 4 + k * 2 + 1];
    xb[((b * C + c) * Hp + hh) * Wp + ww] += g[node * 256 + c];
}

// ---------------------------------------------------------------------------
extern "C" void kernel_launch(void* const* d_in, const int* in_sizes, int n_in,
                              void* d_out, int out_size, void* d_ws, size_t ws_size,
                              hipStream_t stream)
{
    (void)in_sizes; (void)n_in; (void)out_size; (void)ws_size;
    const float* x     = (const float*)d_in[0];
    const float* c1_w  = (const float*)d_in[1];
    const float* bn1_g = (const float*)d_in[2];  const float* bn1_b = (const float*)d_in[3];
    const float* c2_w  = (const float*)d_in[4];
    const float* bn2_g = (const float*)d_in[5];  const float* bn2_b = (const float*)d_in[6];
    const float* c3_w  = (const float*)d_in[7];
    const float* bn3_g = (const float*)d_in[8];  const float* bn3_b = (const float*)d_in[9];
    const float* cb_w  = (const float*)d_in[10];
    const float* bnb_g = (const float*)d_in[11]; const float* bnb_b = (const float*)d_in[12];
    const float* hm1_w = (const float*)d_in[13]; const float* hm1_b = (const float*)d_in[14];
    const float* hm2_w = (const float*)d_in[15]; const float* hm2_b = (const float*)d_in[16];
    const float* hm3_w = (const float*)d_in[17]; const float* hm3_b = (const float*)d_in[18];
    const float* g1_w  = (const float*)d_in[19]; const float* g1_as = (const float*)d_in[20];
    const float* g1_ad = (const float*)d_in[21]; const float* g1_b  = (const float*)d_in[22];
    const float* g2_w  = (const float*)d_in[23]; const float* g2_as = (const float*)d_in[24];
    const float* g2_ad = (const float*)d_in[25]; const float* g2_b  = (const float*)d_in[26];
    const float* u1_w  = (const float*)d_in[27];
    const float* bnu1_g = (const float*)d_in[28]; const float* bnu1_b = (const float*)d_in[29];
    const float* u2_w  = (const float*)d_in[30];
    const float* bnu2_g = (const float*)d_in[31]; const float* bnu2_b = (const float*)d_in[32];
    const float* u3_w  = (const float*)d_in[33];
    const float* bnu3_g = (const float*)d_in[34]; const float* bnu3_b = (const float*)d_in[35];
    const float* pr_w  = (const float*)d_in[36]; const float* pr_b = (const float*)d_in[37];

    // ---- workspace bump allocator (floats) ----
    float* ws = (float*)d_ws;
    size_t off = 0;
    auto alloc = [&](size_t nf) { float* p = ws + off; off += nf; return p; };

    float* x1   = alloc((size_t)8 * 32 * 288 * 512);
    float* p1   = alloc((size_t)8 * 32 * 144 * 256);
    float* x2   = alloc((size_t)8 * 64 * 144 * 256);
    float* p2   = alloc((size_t)8 * 64 * 72 * 128);
    float* x3   = alloc((size_t)8 * 128 * 72 * 128);
    float* p3   = alloc((size_t)8 * 128 * 36 * 64);
    float* xb   = alloc((size_t)8 * 256 * 36 * 64);
    float* h1   = alloc((size_t)8 * 64 * 36 * 64);
    float* h2   = alloc((size_t)8 * 32 * 36 * 64);
    float* heat = alloc((size_t)8 * 36 * 64);
    float* nodes = alloc(144 * 256);
    float* hbuf  = alloc(144 * 256);
    float* g1o   = alloc(144 * 256);
    float* g2o   = alloc(144 * 256);
    float* asb   = alloc(144 * 4);
    float* adb   = alloc(144 * 4);
    float* meanb = alloc(256);
    float* rstdb = alloc(256);
    int*   hwq   = (int*)alloc(64);
    float* d1   = alloc((size_t)8 * 128 * 72 * 128);
    float* d2   = alloc((size_t)8 * 64 * 144 * 256);
    float* d3   = alloc((size_t)8 * 32 * 288 * 512);

    auto conv = [&](const float* in0, int C0, int s0, const float* in1, int C1,
                    const float* wtp, int ks, const float* bias, int act,
                    float* outp, int Nb, int Hh, int wshift, int Cout) {
        int Mtot = Nb * (Hh << wshift);
        dim3 grid((unsigned)((Mtot + 64 * 8 - 1) / (64 * 8)), (Cout + 15) / 16);
        if (ks == 3)
            conv_wmma_kernel<3><<<grid, 256, 0, stream>>>(in0, C0, s0, in1, C1, wtp,
                                                          bias, act, outp, Nb, Hh, wshift, Cout);
        else
            conv_wmma_kernel<1><<<grid, 256, 0, stream>>>(in0, C0, s0, in1, C1, wtp,
                                                          bias, act, outp, Nb, Hh, wshift, Cout);
    };
    auto bn = [&](float* t, int Nb, int C, int Hh, int Ww, const float* g, const float* b) {
        int HW = Hh * Ww;
        bn_stats_kernel<<<C, 256, 0, stream>>>(t, Nb, C, HW, meanb, rstdb);
        dim3 grid((HW + 255) / 256, C, Nb);
        bn_apply_relu_kernel<<<grid, 256, 0, stream>>>(t, C, HW, meanb, rstdb, g, b);
    };
    auto pool = [&](const float* in, float* outp, int NC, int Ho, int woShift) {
        dim3 grid(((Ho << woShift) + 255) / 256, NC);
        maxpool2_kernel<<<grid, 256, 0, stream>>>(in, outp, Ho, woShift);
    };

    // ---- encoder ----  (wshift: 512->9, 256->8, 128->7, 64->6)
    conv(x, 9, 0, nullptr, 0, c1_w, 3, nullptr, 0, x1, 8, 288, 9, 32);
    bn(x1, 8, 32, 288, 512, bn1_g, bn1_b);
    pool(x1, p1, 8 * 32, 144, 8);
    conv(p1, 32, 0, nullptr, 0, c2_w, 3, nullptr, 0, x2, 8, 144, 8, 64);
    bn(x2, 8, 64, 144, 256, bn2_g, bn2_b);
    pool(x2, p2, 8 * 64, 72, 7);
    conv(p2, 64, 0, nullptr, 0, c3_w, 3, nullptr, 0, x3, 8, 72, 7, 128);
    bn(x3, 8, 128, 72, 128, bn3_g, bn3_b);
    pool(x3, p3, 8 * 128, 36, 6);
    conv(p3, 128, 0, nullptr, 0, cb_w, 3, nullptr, 0, xb, 8, 36, 6, 256);
    bn(xb, 8, 256, 36, 64, bnb_g, bnb_b);

    // ---- heatmap head + top-2 ROIs ----
    conv(xb, 256, 0, nullptr, 0, hm1_w, 3, hm1_b, 1, h1, 8, 36, 6, 64);
    conv(h1, 64, 0, nullptr, 0, hm2_w, 3, hm2_b, 1, h2, 8, 36, 6, 32);
    conv(h2, 32, 0, nullptr, 0, hm3_w, 1, hm3_b, 2, heat, 8, 36, 6, 1);
    top2_kernel<<<8, 256, 0, stream>>>(heat, 36 * 64, 64, hwq);
    gather_nodes_kernel<<<144, 256, 0, stream>>>(xb, hwq, nodes, 256, 36, 64);

    // ---- GAT layer 1 ----
    gemm_nt_wmma_kernel<<<dim3(2, 16), 256, 0, stream>>>(nodes, g1_w, hbuf, 144, 256, 256);
    gat_coef_kernel<<<3, 256, 0, stream>>>(hbuf, g1_as, g1_ad, asb, adb);
    gat_aggregate_kernel<<<144, 256, 0, stream>>>(hbuf, asb, adb, g1_b, g1o);
    // ---- GAT layer 2 ----
    gemm_nt_wmma_kernel<<<dim3(2, 16), 256, 0, stream>>>(g1o, g2_w, hbuf, 144, 256, 256);
    gat_coef_kernel<<<3, 256, 0, stream>>>(hbuf, g2_as, g2_ad, asb, adb);
    gat_aggregate_kernel<<<144, 256, 0, stream>>>(hbuf, asb, adb, g2_b, g2o);
    // scatter last-frame node features back into bottleneck
    scatter_add_kernel<<<16, 256, 0, stream>>>(g2o, hwq, xb, 256, 36, 64);

    // ---- decoder (implicit upsample + concat inside conv gather) ----
    conv(xb, 256, 1, x3, 128, u1_w, 3, nullptr, 0, d1, 8, 72, 7, 128);
    bn(d1, 8, 128, 72, 128, bnu1_g, bnu1_b);
    conv(d1, 128, 1, x2, 64, u2_w, 3, nullptr, 0, d2, 8, 144, 8, 64);
    bn(d2, 8, 64, 144, 256, bnu2_g, bnu2_b);
    conv(d2, 64, 1, x1, 32, u3_w, 3, nullptr, 0, d3, 8, 288, 9, 32);
    bn(d3, 8, 32, 288, 512, bnu3_g, bnu3_b);
    // final 1x1 conv + bias + sigmoid -> output (8,9,288,512)
    conv(d3, 32, 0, nullptr, 0, pr_w, 1, pr_b, 2, (float*)d_out, 8, 288, 9, 9);
}
